// Model_42116449304608
// MI455X (gfx1250) — compile-verified, matrix operand order checked by reading, not run on hardware
//
#include <hip/hip_runtime.h>
#include <hip/hip_bf16.h>
#include <math.h>

// ---------------- model constants (longformer-base-4096, B=2, S=2048) ------
#define Lz   12
#define NHh  12
#define Dm   768
#define DHh  64
#define FFm  3072
#define WINz 256
#define Bn   2
#define Sn   2048
#define BSn  (Bn*Sn)
#define NBLK (Sn/WINz)   // 8

// ---------------- WMMA plumbing -------------------------------------------
typedef _Float16 v16h  __attribute__((ext_vector_type(16)));
typedef _Float16 half8 __attribute__((ext_vector_type(8)));
typedef float    v8f   __attribute__((ext_vector_type(8)));

__device__ __forceinline__ v8f wmma32(v16h a, v16h b, v8f c) {
    // (neg_a, A, neg_b, B, c_mod, C, reuse_a, reuse_b)
    return __builtin_amdgcn_wmma_f32_16x16x32_f16(false, a, false, b,
                                                  (short)0, c, false, false);
}

// Load one 16x32 f16 fragment row for this lane from LDS.
// Per CDNA5 ISA 16-bit A layout: lane L (m = L%16..) holds two contiguous
// 8-element K runs: [8*half, 8*half+8) and [16+8*half, 16+8*half+8).
// `p` must point at (row_base + kbase) in an LDS row whose stride keeps
// 16-byte alignment (stride % 8 == 0 elements).
__device__ __forceinline__ v16h frag_ld(const _Float16* p, int hf) {
    half8 lo = *(const half8*)(p + 8 * hf);
    half8 hi = *(const half8*)(p + 16 + 8 * hf);
    return __builtin_shufflevector(lo, hi, 0,1,2,3,4,5,6,7,8,9,10,11,12,13,14,15);
}

__device__ __forceinline__ float gelu_exact(float x) {
    return 0.5f * x * (1.0f + erff(x * 0.70710678118654752f));
}

// ---------------- generic f32-in / f16-WMMA / f32-out GEMM ------------------
// C[M,N] = act(A[M,K] @ B[K,N] + bias).  M%64==0, N%64==0, K%32==0.
// grid = (N/64, M/64), block = 256 (8 waves).
__global__ __launch_bounds__(256) void gemm_kernel(
    const float* __restrict__ A, const float* __restrict__ Bw,
    const float* __restrict__ bias, float* __restrict__ C,
    int M, int N, int K, int act)
{
    __shared__ __align__(16) _Float16 As[64 * 40];   // [m][k], stride 40
    __shared__ __align__(16) _Float16 Bs[64 * 40];   // [n][k] (B transposed)

    const int tid  = threadIdx.x;
    const int lane = tid & 31;
    const int w    = tid >> 5;
    const int wr   = w >> 2;          // 0..1 : row half of tile
    const int wc   = w & 3;           // 0..3 : 16-col strip
    const int mr   = lane & 15;
    const int hf   = lane >> 4;

    const int m0 = blockIdx.y * 64;
    const int n0 = blockIdx.x * 64;

    v8f acc0 = {}, acc1 = {};

    const int ar = tid >> 2;          // 0..63 A row
    const int ak = (tid & 3) * 8;     // 0,8,16,24
    const int bn = tid & 63;          // 0..63 B col
    const int bk = (tid >> 6) * 8;    // 0,8,16,24

    for (int kk = 0; kk < K; kk += 32) {
        const float* ap = A + (size_t)(m0 + ar) * K + kk + ak;
        #pragma unroll
        for (int i = 0; i < 8; ++i)
            As[ar * 40 + ak + i] = (_Float16)ap[i];
        #pragma unroll
        for (int i = 0; i < 8; ++i)
            Bs[bn * 40 + bk + i] =
                (_Float16)Bw[(size_t)(kk + bk + i) * N + n0 + bn];
        __syncthreads();

        v16h a0 = frag_ld(&As[((wr * 2 + 0) * 16 + mr) * 40], hf);
        v16h a1 = frag_ld(&As[((wr * 2 + 1) * 16 + mr) * 40], hf);
        v16h bf = frag_ld(&Bs[(wc * 16 + mr) * 40], hf);
        acc0 = wmma32(a0, bf, acc0);
        acc1 = wmma32(a1, bf, acc1);
        __syncthreads();
    }

    const int n  = n0 + wc * 16 + mr;
    const float bv = bias ? bias[n] : 0.0f;
    #pragma unroll
    for (int r = 0; r < 8; ++r) {
        int ma = m0 + wr * 32 + r + 8 * hf;       // tile rows [0,16)
        int mb = ma + 16;                         // tile rows [16,32)
        float va = acc0[r] + bv;
        float vb = acc1[r] + bv;
        if (act) { va = gelu_exact(va); vb = gelu_exact(vb); }
        C[(size_t)ma * N + n] = va;
        C[(size_t)mb * N + n] = vb;
    }
}

// ---------------- flash-style banded attention ------------------------------
// One block per (b, head, seq-block of 256, 64-query subtile).
// Q,K,V,O are [B*S, D] with head slice at column h*64.
__global__ __launch_bounds__(256) void attn_kernel(
    const float* __restrict__ Q, const float* __restrict__ K,
    const float* __restrict__ V, const int* __restrict__ amask,
    float* __restrict__ O)
{
    __shared__ __align__(16) _Float16 Qs [64 * 72];  // [q][dh]
    __shared__ __align__(16) _Float16 KVs[64 * 72];  // K:[key][dh]  V:[dh][key]
    __shared__ __align__(16) _Float16 Ps [64 * 72];  // [q][key]
    __shared__ float Ss[64 * 65];                    // raw scores chunk
    __shared__ float rowm[64], rowl[64], rowsc[64];

    const int tid  = threadIdx.x;
    const int lane = tid & 31;
    const int w    = tid >> 5;
    const int mi   = w & 3;           // 16-row tile   (0..3)
    const int nc   = w >> 2;          // col-tile pair (0..1)
    const int mr   = lane & 15;
    const int hf   = lane >> 4;

    int idx  = blockIdx.x;
    const int qt  = idx & 3;  idx >>= 2;
    const int nbk = idx % NBLK; idx /= NBLK;
    const int hh  = idx % NHh;
    const int bb  = idx / NHh;

    const int qbase = bb * Sn + nbk * WINz + qt * 64;  // first query row (global)
    const float scl = 0.125f;                          // 1/sqrt(64)

    for (int e = tid; e < 64 * 64; e += 256) {
        int i = e >> 6, kd = e & 63;
        Qs[i * 72 + kd] =
            (_Float16)(Q[(size_t)(qbase + i) * Dm + hh * DHh + kd] * scl);
    }
    if (tid < 64) { rowm[tid] = -1e30f; rowl[tid] = 0.0f; }
    v8f o0 = {}, o1 = {};
    __syncthreads();

    for (int c = 0; c < 12; ++c) {              // 12 x 64 = 768-key window
        const int kw0 = (nbk - 1) * WINz + c * 64;  // global index of chunk key 0

        // ---- stage K chunk as [key][dh] ----
        for (int e = tid; e < 64 * 64; e += 256) {
            int n = e >> 6, kd = e & 63;
            int gk = kw0 + n;
            float val = 0.0f;
            if (gk >= 0 && gk < Sn)
                val = K[(size_t)(bb * Sn + gk) * Dm + hh * DHh + kd];
            KVs[n * 72 + kd] = (_Float16)val;
        }
        __syncthreads();

        // ---- S = Q Kᵀ (64x64) ----
        v8f s0 = {}, s1 = {};
        #pragma unroll
        for (int kk = 0; kk < 64; kk += 32) {
            v16h af = frag_ld(&Qs[(mi * 16 + mr) * 72 + kk], hf);
            v16h b0 = frag_ld(&KVs[((nc * 2 + 0) * 16 + mr) * 72 + kk], hf);
            v16h b1 = frag_ld(&KVs[((nc * 2 + 1) * 16 + mr) * 72 + kk], hf);
            s0 = wmma32(af, b0, s0);
            s1 = wmma32(af, b1, s1);
        }

        // ---- scatter scores with band + padding mask ----
        #pragma unroll
        for (int r = 0; r < 8; ++r) {
            int ml = mi * 16 + r + 8 * hf;     // local query 0..63
            int p  = qt * 64 + ml;             // pos within 256-block
            #pragma unroll
            for (int half_tile = 0; half_tile < 2; ++half_tile) {
                int nl = (nc * 2 + half_tile) * 16 + mr;   // local key 0..63
                int t  = c * 64 + nl;                      // window coord 0..767
                int gk = kw0 + nl;
                int d  = t - WINz - p;
                bool ok = (d >= -WINz) && (d <= WINz) &&
                          (gk >= 0) && (gk < Sn) &&
                          (amask[bb * Sn + gk] > 0);
                float sv = half_tile ? s1[r] : s0[r];
                Ss[ml * 65 + nl] = ok ? sv : -1e9f;
            }
        }
        __syncthreads();

        // ---- online softmax bookkeeping (one thread per row) ----
        if (tid < 64) {
            float mprev = rowm[tid];
            float mx = mprev;
            for (int j = 0; j < 64; ++j) mx = fmaxf(mx, Ss[tid * 65 + j]);
            float sc = expf(mprev - mx);
            float l = rowl[tid] * sc;
            for (int j = 0; j < 64; ++j) {
                float pv = expf(Ss[tid * 65 + j] - mx);
                Ps[tid * 72 + j] = (_Float16)pv;
                l += pv;
            }
            rowsc[tid] = sc; rowm[tid] = mx; rowl[tid] = l;
        }
        __syncthreads();

        // ---- rescale running O ----
        #pragma unroll
        for (int r = 0; r < 8; ++r) {
            float sc = rowsc[mi * 16 + r + 8 * hf];
            o0[r] *= sc; o1[r] *= sc;
        }

        // ---- stage V chunk transposed: [dh][key] ----
        for (int e = tid; e < 64 * 64; e += 256) {
            int nd = e >> 6, kl = e & 63;
            int gk = kw0 + kl;
            float val = 0.0f;
            if (gk >= 0 && gk < Sn)
                val = V[(size_t)(bb * Sn + gk) * Dm + hh * DHh + nd];
            KVs[nd * 72 + kl] = (_Float16)val;
        }
        __syncthreads();

        // ---- O += P V ----
        #pragma unroll
        for (int kk = 0; kk < 64; kk += 32) {
            v16h af = frag_ld(&Ps[(mi * 16 + mr) * 72 + kk], hf);
            v16h b0 = frag_ld(&KVs[((nc * 2 + 0) * 16 + mr) * 72 + kk], hf);
            v16h b1 = frag_ld(&KVs[((nc * 2 + 1) * 16 + mr) * 72 + kk], hf);
            o0 = wmma32(af, b0, o0);
            o1 = wmma32(af, b1, o1);
        }
        __syncthreads();
    }

    #pragma unroll
    for (int r = 0; r < 8; ++r) {
        int ml = mi * 16 + r + 8 * hf;
        float inv = 1.0f / rowl[ml];
        size_t row = (size_t)(qbase + ml) * Dm + hh * DHh;
        O[row + (nc * 2 + 0) * 16 + mr] = o0[r] * inv;
        O[row + (nc * 2 + 1) * 16 + mr] = o1[r] * inv;
    }
}

// ---------------- residual + LayerNorm -------------------------------------
__global__ __launch_bounds__(256) void ln_kernel(
    const float* __restrict__ x, const float* __restrict__ res,
    const float* __restrict__ g, const float* __restrict__ bta,
    float* __restrict__ out, int d)
{
    __shared__ float r1[256], r2[256];
    const int row = blockIdx.x, tid = threadIdx.x;
    const float* xp = x + (size_t)row * d;
    const float* rp = res ? res + (size_t)row * d : nullptr;
    float s = 0.f, s2 = 0.f;
    for (int i = tid; i < d; i += 256) {
        float v = xp[i] + (rp ? rp[i] : 0.f);
        s += v; s2 += v * v;
    }
    r1[tid] = s; r2[tid] = s2; __syncthreads();
    for (int off = 128; off > 0; off >>= 1) {
        if (tid < off) { r1[tid] += r1[tid + off]; r2[tid] += r2[tid + off]; }
        __syncthreads();
    }
    float mean = r1[0] / d;
    float var  = r2[0] / d - mean * mean;
    float rstd = rsqrtf(var + 1e-5f);
    for (int i = tid; i < d; i += 256) {
        float v = xp[i] + (rp ? rp[i] : 0.f);
        out[(size_t)row * d + i] = (v - mean) * rstd * g[i] + bta[i];
    }
}

// ---------------- embeddings -----------------------------------------------
__global__ void posids_kernel(const int* __restrict__ ids, int* __restrict__ pids)
{
    int b = blockIdx.x;
    if (threadIdx.x != 0) return;
    int c = 0;
    for (int s = 0; s < Sn; ++s) {
        int ne = (ids[b * Sn + s] != 1);
        c += ne;
        pids[b * Sn + s] = c * ne + 1;
    }
}

__global__ void embed_kernel(const int* __restrict__ ids, const int* __restrict__ pids,
                             const float* __restrict__ word, const float* __restrict__ pos,
                             const float* __restrict__ tok, float* __restrict__ out)
{
    size_t i = (size_t)blockIdx.x * blockDim.x + threadIdx.x;
    if (i >= (size_t)BSn * Dm) return;
    int row = (int)(i / Dm), dd = (int)(i % Dm);
    out[i] = word[(size_t)ids[row] * Dm + dd] + pos[(size_t)pids[row] * Dm + dd] + tok[dd];
}

// ---------------- pooler / classifier --------------------------------------
__global__ void pooler_kernel(const float* __restrict__ h, const float* __restrict__ Wp,
                              const float* __restrict__ bp, float* __restrict__ pooled)
{
    int j = blockIdx.x * blockDim.x + threadIdx.x;
    if (j >= Bn * Dm) return;
    int b = j / Dm, col = j % Dm;
    float s = bp[col];
    const float* hrow = h + (size_t)(b * Sn) * Dm;   // token 0 of batch b
    for (int k = 0; k < Dm; ++k) s += hrow[k] * Wp[(size_t)k * Dm + col];
    pooled[j] = tanhf(s);
}

__global__ void cls_kernel(const float* __restrict__ pooled, const float* __restrict__ Wc,
                           const float* __restrict__ bc, float* __restrict__ out, int C)
{
    int j = blockIdx.x * blockDim.x + threadIdx.x;
    if (j >= Bn * C) return;
    int b = j / C, c = j % C;
    float s = bc[c];
    for (int k = 0; k < Dm; ++k) s += pooled[b * Dm + k] * Wc[(size_t)k * C + c];
    out[j] = 1.0f / (1.0f + expf(-s));
}

// ---------------- driver ----------------------------------------------------
extern "C" void kernel_launch(void* const* d_in, const int* in_sizes, int n_in,
                              void* d_out, int out_size, void* d_ws, size_t ws_size,
                              hipStream_t stream)
{
    (void)in_sizes; (void)n_in; (void)out_size; (void)ws_size;

    const int*   ids   = (const int*)  d_in[0];
    const int*   amask = (const int*)  d_in[1];
    const float* word  = (const float*)d_in[2];
    const float* pos   = (const float*)d_in[3];
    const float* tok   = (const float*)d_in[4];
    const float* elng  = (const float*)d_in[5];
    const float* elnb  = (const float*)d_in[6];
    const float* Wq    = (const float*)d_in[7];
    const float* bq    = (const float*)d_in[8];
    const float* Wk    = (const float*)d_in[9];
    const float* bk    = (const float*)d_in[10];
    const float* Wv    = (const float*)d_in[11];
    const float* bv    = (const float*)d_in[12];
    const float* Wo    = (const float*)d_in[13];
    const float* bo    = (const float*)d_in[14];
    const float* ln1g  = (const float*)d_in[15];
    const float* ln1b  = (const float*)d_in[16];
    const float* Wf1   = (const float*)d_in[17];
    const float* bf1   = (const float*)d_in[18];
    const float* Wf2   = (const float*)d_in[19];
    const float* bf2   = (const float*)d_in[20];
    const float* ln2g  = (const float*)d_in[21];
    const float* ln2b  = (const float*)d_in[22];
    const float* Wp    = (const float*)d_in[23];
    const float* bp    = (const float*)d_in[24];
    const float* Wc    = (const float*)d_in[25];
    const float* bc    = (const float*)d_in[26];

    const size_t HS = (size_t)BSn * Dm;          // 3,145,728 floats
    float* ws     = (float*)d_ws;
    float* h      = ws;
    float* q      = h    + HS;
    float* kbuf   = q    + HS;
    float* vbuf   = kbuf + HS;
    float* att    = vbuf + HS;
    float* tmp    = att  + HS;
    float* f1     = tmp  + HS;                   // B*S*FF
    float* pooled = f1   + (size_t)BSn * FFm;
    int*   pids   = (int*)(pooled + Bn * Dm);

    posids_kernel<<<Bn, 32, 0, stream>>>(ids, pids);
    embed_kernel<<<(int)((HS + 255) / 256), 256, 0, stream>>>(ids, pids, word, pos, tok, tmp);
    ln_kernel<<<BSn, 256, 0, stream>>>(tmp, nullptr, elng, elnb, h, Dm);

    dim3 gD(Dm / 64, BSn / 64);   // 12 x 64
    dim3 gF(FFm / 64, BSn / 64);  // 48 x 64
    const int attn_blocks = Bn * NHh * NBLK * 4;  // 768

    for (int l = 0; l < Lz; ++l) {
        const size_t od = (size_t)l * Dm * Dm;
        const size_t of = (size_t)l * Dm * FFm;
        gemm_kernel<<<gD, 256, 0, stream>>>(h, Wq + od, bq + l * Dm, q,    BSn, Dm, Dm, 0);
        gemm_kernel<<<gD, 256, 0, stream>>>(h, Wk + od, bk + l * Dm, kbuf, BSn, Dm, Dm, 0);
        gemm_kernel<<<gD, 256, 0, stream>>>(h, Wv + od, bv + l * Dm, vbuf, BSn, Dm, Dm, 0);
        attn_kernel<<<attn_blocks, 256, 0, stream>>>(q, kbuf, vbuf, amask, att);
        gemm_kernel<<<gD, 256, 0, stream>>>(att, Wo + od, bo + l * Dm, tmp, BSn, Dm, Dm, 0);
        ln_kernel<<<BSn, 256, 0, stream>>>(tmp, h, ln1g + l * Dm, ln1b + l * Dm, h, Dm);
        gemm_kernel<<<gF, 256, 0, stream>>>(h, Wf1 + of, bf1 + l * FFm, f1, BSn, FFm, Dm, 1);
        gemm_kernel<<<gD, 256, 0, stream>>>(f1, Wf2 + of, bf2 + l * Dm, tmp, BSn, Dm, FFm, 0);
        ln_kernel<<<BSn, 256, 0, stream>>>(tmp, h, ln2g + l * Dm, ln2b + l * Dm, h, Dm);
    }

    pooler_kernel<<<(Bn * Dm + 255) / 256, 256, 0, stream>>>(h, Wp, bp, pooled);
    cls_kernel<<<1, 64, 0, stream>>>(pooled, Wc, bc, (float*)d_out, 16);
}
